// GatedSpikingReservoirStep_37572373905508
// MI455X (gfx1250) — matrix operand order repaired and checked by reading, not compiled, not run on hardware
//
#include <hip/hip_runtime.h>

// ---------------------------------------------------------------------------
// GatedSpikingReservoirStep for MI455X (gfx1250, wave32, WMMA bf16)
//
//   S_pre = X @ Wi          (2048x1024 @ 1024x4096)
//   G_pre = X @ Wg          (2048x1024 @ 1024x4096)
//   R_pre = S @ Wr          (2048x4096 @ 4096x4096)  + fused epilogue:
//   ns  = (0.9*S + 0.1*tanh(S_pre + R_pre)) * sigmoid(G_pre)
//   out = ns > 0.5 ? 1 : 0
//
// All operands (143 MB f32) are L2-resident (192 MB L2). Tiles are staged
// f32 -> bf16 into LDS and multiplied with v_wmma_f32_16x16x32_bf16.
// Register-staged pipeline: global loads of tile i+1 overlap WMMAs of tile i.
// d_out's two halves double as scratch for S_pre / G_pre.
// ---------------------------------------------------------------------------

typedef __attribute__((ext_vector_type(16))) __bf16 bf16x16;
typedef __attribute__((ext_vector_type(8)))  float  f32x8;

union Frag16 { bf16x16 v; uint4 u[2]; };

#define TILE_M 128
#define TILE_N 128
#define TILE_K 64
#define LDSL   72   // u16 elements per LDS row: 64 + 8 pad (144B = 9*16B)

// Pack two f32 -> packed bf16 pair (lo in [15:0]) with round-to-nearest-even.
// v_perm_b32 grabs the two rounded high halves in one op (avoids v_mov_b16
// chains the shift/or form lowered to).
__device__ __forceinline__ unsigned int pack2(float lo, float hi) {
  unsigned int ul = __float_as_uint(lo);
  unsigned int uh = __float_as_uint(hi);
  ul += 0x7FFFu + ((ul >> 16) & 1u);   // RNE bias
  uh += 0x7FFFu + ((uh >> 16) & 1u);
#if __has_builtin(__builtin_amdgcn_perm)
  // {uh,ul} bytes: result = [uh.b3, uh.b2, ul.b3, ul.b2]
  return __builtin_amdgcn_perm(uh, ul, 0x07060302u);
#else
  return (ul >> 16) | (uh & 0xFFFF0000u);
#endif
}

// A fragment, 16x32 bf16 (ISA 7.12.2): lanes 0-15 row M=lane, v0-3 = K0..7,
// v4-7 = K16..23; lanes 16-31 same rows, K8..15 / K24..31. ks = k-slab (0/32).
__device__ __forceinline__ bf16x16 loadA(const unsigned short* As, int rowBase,
                                         int lane, int ks) {
  const int lr = lane & 15, hi = (lane >> 4) * 8;
  const unsigned short* p = As + (rowBase + lr) * LDSL + ks;
  Frag16 f;
  f.u[0] = *(const uint4*)(p + hi);        // K  (hi? 8:0) .. +7
  f.u[1] = *(const uint4*)(p + 16 + hi);   // K 16+(hi? 8:0) .. +7
  return f.v;
}

// B fragment, 32x16 bf16: lane = N (mod 16); lanes 0-15 hold K0..7 / K16..23,
// lanes 16-31 hold K8..15 / K24..31.  Bs is stored transposed: Bs[n][k].
__device__ __forceinline__ bf16x16 loadB(const unsigned short* Bs, int colBase,
                                         int lane, int ks) {
  const int ln = lane & 15, half = (lane >> 4) * 8;
  const unsigned short* p = Bs + (colBase + ln) * LDSL + ks + half;
  Frag16 f;
  f.u[0] = *(const uint4*)(p);
  f.u[1] = *(const uint4*)(p + 16);
  return f.v;
}

// MODE 0: store raw accumulator to C0.
// MODE 1: fused epilogue: read S_pre from nsBuf, G_pre from outBuf, state
//         elementwise; write final new_state / spike outputs.
template <int MODE>
__global__ __launch_bounds__(256) void reservoir_gemm(
    const float* __restrict__ A, const float* __restrict__ W,
    int M, int K, int N,
    float* __restrict__ C0,
    const float* __restrict__ state,
    float* __restrict__ outBuf, float* __restrict__ nsBuf) {

  __shared__ __align__(16) unsigned short As[TILE_M * LDSL];
  __shared__ __align__(16) unsigned short Bs[TILE_N * LDSL];

  const int tid  = threadIdx.x;
  const int lane = tid & 31;
  const int wave = tid >> 5;          // 8 waves
  const int wRow = (wave >> 1) * 32;  // 4 wave-rows of 32
  const int wCol = (wave & 1) * 64;   // 2 wave-cols of 64

  const int mBase = blockIdx.y * TILE_M;
  const int nBase = blockIdx.x * TILE_N;

  f32x8 acc[2][4];
#pragma unroll
  for (int i = 0; i < 2; ++i)
#pragma unroll
    for (int j = 0; j < 4; ++j)
#pragma unroll
      for (int e = 0; e < 8; ++e) acc[i][j][e] = 0.0f;

  // A staging map: 128x64 f32 tile, 8 passes of 16 rows, float4 per thread
  const int kqA = tid & 15;   // float4 column: k = kqA*4
  const int r0A = tid >> 4;   // row within pass
  // B staging map: 64x128 f32 tile, k-pairs so LDS stores are packed b32
  const int nqB = tid & 31;   // n = nqB*4
  const int kpB = tid >> 5;   // k-pair row within pass (0..7)

  float4 ra[8];               // A tile staged in regs (pipeline)
  float4 rb0[4], rb1[4];      // B tile: two consecutive k rows per pass

  // ---- prologue: fetch tile kb=0 into registers ----
  {
#pragma unroll
    for (int p = 0; p < 8; ++p)
      ra[p] = *(const float4*)(A + (mBase + p * 16 + r0A) * K + kqA * 4);
#pragma unroll
    for (int p = 0; p < 4; ++p) {
      const int k0 = (p * 8 + kpB) * 2;
      rb0[p] = *(const float4*)(W + (k0    ) * N + nBase + nqB * 4);
      rb1[p] = *(const float4*)(W + (k0 + 1) * N + nBase + nqB * 4);
    }
  }

#pragma unroll 1
  for (int kb = 0; kb < K; kb += TILE_K) {
    // ---- convert + store staged registers to LDS ----
#pragma unroll
    for (int p = 0; p < 8; ++p) {
      uint2 pk; pk.x = pack2(ra[p].x, ra[p].y); pk.y = pack2(ra[p].z, ra[p].w);
      *(uint2*)&As[(p * 16 + r0A) * LDSL + kqA * 4] = pk;
    }
#pragma unroll
    for (int p = 0; p < 4; ++p) {
      const int k0 = (p * 8 + kpB) * 2;
      const int n  = nqB * 4;
      *(unsigned int*)&Bs[(n + 0) * LDSL + k0] = pack2(rb0[p].x, rb1[p].x);
      *(unsigned int*)&Bs[(n + 1) * LDSL + k0] = pack2(rb0[p].y, rb1[p].y);
      *(unsigned int*)&Bs[(n + 2) * LDSL + k0] = pack2(rb0[p].z, rb1[p].z);
      *(unsigned int*)&Bs[(n + 3) * LDSL + k0] = pack2(rb0[p].w, rb1[p].w);
    }
    __syncthreads();

    // ---- issue global fetch of next tile (overlaps WMMAs below) ----
    const int kn = kb + TILE_K;
    if (kn < K) {
#pragma unroll
      for (int p = 0; p < 8; ++p)
        ra[p] = *(const float4*)(A + (mBase + p * 16 + r0A) * K + kn + kqA * 4);
#pragma unroll
      for (int p = 0; p < 4; ++p) {
        const int k0 = kn + (p * 8 + kpB) * 2;
        rb0[p] = *(const float4*)(W + (k0    ) * N + nBase + nqB * 4);
        rb1[p] = *(const float4*)(W + (k0 + 1) * N + nBase + nqB * 4);
      }
    }

    // ---- 2 k-slabs x 8 WMMAs from LDS ----
#pragma unroll
    for (int s = 0; s < 2; ++s) {
      const int ks = s * 32;
      bf16x16 a[2], b[4];
#pragma unroll
      for (int i = 0; i < 2; ++i) a[i] = loadA(As, wRow + i * 16, lane, ks);
#pragma unroll
      for (int j = 0; j < 4; ++j) b[j] = loadB(Bs, wCol + j * 16, lane, ks);
#pragma unroll
      for (int i = 0; i < 2; ++i)
#pragma unroll
        for (int j = 0; j < 4; ++j)
          acc[i][j] = __builtin_amdgcn_wmma_f32_16x16x32_bf16(
              false, a[i], false, b[j], (short)0, acc[i][j], false, false);
    }
    __syncthreads();
  }

  // ---- epilogue: C layout = VGPR r -> M = r + (lane>=16 ? 8:0), N = lane&15.
  // 32-bit index math (max 2048*4096 < 2^31) so stores fold into imm offsets.
  const int laneN   = lane & 15;
  const int laneMof = (lane >> 4) * 8;
#pragma unroll
  for (int i = 0; i < 2; ++i) {
#pragma unroll
    for (int j = 0; j < 4; ++j) {
      const int base = (mBase + wRow + i * 16 + laneMof) * N
                     + (nBase + wCol + j * 16 + laneN);
#pragma unroll
      for (int r = 0; r < 8; ++r) {
        const int idx = base + r * N;
        const float v = acc[i][j][r];
        if (MODE == 0) {
          C0[idx] = v;
        } else {
          const float pre = nsBuf[idx] + v;                  // X@Wi + S@Wr
          const float gl  = outBuf[idx];                     // X@Wg
          const float g   = 1.0f / (1.0f + __expf(-gl));     // sigmoid
          const float st  = state[idx];
          const float ns  = (0.9f * st + 0.1f * tanhf(pre)) * g;
          nsBuf[idx]  = ns;                                  // new_state
          outBuf[idx] = ns > 0.5f ? 1.0f : 0.0f;             // spikes
        }
      }
    }
  }
}

extern "C" void kernel_launch(void* const* d_in, const int* in_sizes, int n_in,
                              void* d_out, int out_size, void* d_ws, size_t ws_size,
                              hipStream_t stream) {
  (void)in_sizes; (void)n_in; (void)d_ws; (void)ws_size; (void)out_size;
  const float* inputs = (const float*)d_in[0];
  // d_in[1] = prev_output: unused by the reference
  const float* state  = (const float*)d_in[2];
  const float* Wi     = (const float*)d_in[3];
  const float* Wr     = (const float*)d_in[4];
  const float* Wg     = (const float*)d_in[5];

  const int M = 2048, Din = 1024, R = 4096;
  float* outBuf = (float*)d_out;                   // spikes   (also G_pre scratch)
  float* nsBuf  = (float*)d_out + (size_t)M * R;   // new_state (also S_pre scratch)

  dim3 block(256);
  dim3 grid(R / TILE_N, M / TILE_M);  // 32 x 16 blocks

  reservoir_gemm<0><<<grid, block, 0, stream>>>(inputs, Wi, M, Din, R,
                                                nsBuf, nullptr, nullptr, nullptr);
  reservoir_gemm<0><<<grid, block, 0, stream>>>(inputs, Wg, M, Din, R,
                                                outBuf, nullptr, nullptr, nullptr);
  reservoir_gemm<1><<<grid, block, 0, stream>>>(state, Wr, M, R, R,
                                                nullptr, state, outBuf, nsBuf);
}
